// SoftDTWKDLoss_73323681677450
// MI455X (gfx1250) — matrix operand order repaired and checked by reading, not compiled, not run on hardware
//
#include <hip/hip_runtime.h>

// ---------------------------------------------------------------------------
// SoftDTW KD loss for MI455X (gfx1250, wave32).
//   B=64, LS=LT=256, D=1024, gamma=1.0
//   Stage 1: fused f32->bf16 convert (RNE) + squared row norms (reads inputs once)
//   Stage 2: dist = sqrt(max(s2+t2-2*S@T^T, 1e-12)) via v_wmma_f32_16x16x32_bf16
//            operands loaded as pre-packed bf16 -> pure b128 loads, no VALU packing
//   Stage 3: soft-DTW anti-diagonal wavefront, 1 workgroup / batch
//   Stage 4: mean over batches
// ---------------------------------------------------------------------------

typedef __attribute__((ext_vector_type(16))) __bf16 v16bf;
typedef __attribute__((ext_vector_type(8)))  float  v8f;

#define BATCH 64
#define LSEQ  256
#define DDIM  1024

__device__ __forceinline__ unsigned int pack_bf16_2(float lo, float hi) {
    // round-to-nearest-even f32 -> bf16, packed pair (used only in Stage 1)
    unsigned int ul = __float_as_uint(lo);
    unsigned int uh = __float_as_uint(hi);
    ul = (ul + 0x7FFFu + ((ul >> 16) & 1u)) >> 16;
    uh = (uh + 0x7FFFu + ((uh >> 16) & 1u)) >> 16;
    return (ul & 0xFFFFu) | (uh << 16);
}

// ---------------------------------------------------------------------------
// Stage 1: one wave per row. Convert row to bf16 (memory-bound, done once)
// and compute the squared row norm in f32.
// Rows [0, B*L)    -> student -> Sb, s2
// Rows [B*L, 2B*L) -> teacher -> Tb, t2
// ---------------------------------------------------------------------------
__global__ __launch_bounds__(256) void convert_norms_kernel(
    const float* __restrict__ S, const float* __restrict__ T,
    unsigned short* __restrict__ Sb, unsigned short* __restrict__ Tb,
    float* __restrict__ s2, float* __restrict__ t2) {
    const int wave = threadIdx.x >> 5;
    const int lane = threadIdx.x & 31;
    const int row  = blockIdx.x * 8 + wave;            // 0 .. 2*B*L-1
    const float* src;
    unsigned short* dstb;
    float* dstn;
    if (row < BATCH * LSEQ) {
        src  = S + (size_t)row * DDIM;
        dstb = Sb + (size_t)row * DDIM;
        dstn = s2 + row;
    } else {
        const int r = row - BATCH * LSEQ;
        src  = T + (size_t)r * DDIM;
        dstb = Tb + (size_t)r * DDIM;
        dstn = t2 + r;
    }
    float acc = 0.0f;
    #pragma unroll
    for (int c = lane * 4; c < DDIM; c += 128) {
        float4 v = *reinterpret_cast<const float4*>(src + c);
        acc += v.x * v.x + v.y * v.y + v.z * v.z + v.w * v.w;
        uint2 p;
        p.x = pack_bf16_2(v.x, v.y);
        p.y = pack_bf16_2(v.z, v.w);
        *reinterpret_cast<uint2*>(dstb + c) = p;       // 4 bf16 = 8 bytes
    }
    #pragma unroll
    for (int off = 16; off >= 1; off >>= 1)
        acc += __shfl_down(acc, off, 32);
    if (lane == 0) *dstn = acc;
}

// ---------------------------------------------------------------------------
// Stage 2: batched S @ T^T via WMMA bf16, fused distance epilogue.
// Grid: (4, BATCH), 256 threads = 8 waves. Each wave -> one 16(M) x 128(N)
// strip of C; 32 strips per batch. Operands are pre-packed bf16, so each
// WMMA operand is assembled from b128 loads only:
//
// A tile (16x32 bf16, doc layout): lane = {half=L>>4, m=L&15}
//   dwords 0-3 <- K = 8*half + 0..7   (one uint4 load)
//   dwords 4-7 <- K = 16+8*half+0..7  (one uint4 load)
// B tile (32x16 bf16, B[k][n] = T[n][k]): lane = {half, n=L&15}
//   dwords 0-3 <- K = 8*half + 0..7, dwords 4-7 <- K = 16+8*half+0..7 (two uint4 loads)
// C/D layout: lane L -> n = L&15; VGPR r -> m = r + 8*(L>>4).
// ---------------------------------------------------------------------------
__global__ __launch_bounds__(256) void dist_wmma_kernel(
    const unsigned short* __restrict__ Sb, const unsigned short* __restrict__ Tb,
    const float* __restrict__ s2, const float* __restrict__ t2,
    float* __restrict__ dist) {
    const int b    = blockIdx.y;
    const int wave = threadIdx.x >> 5;
    const int lane = threadIdx.x & 31;
    const int sid  = blockIdx.x * 8 + wave;   // 0..31
    const int m16    = sid & 15;              // M tile 0..15
    const int nstrip = sid >> 4;              // N strip 0..1 (128 cols each)
    const int half = lane >> 4;
    const int l15  = lane & 15;

    const unsigned short* Srow = Sb + ((size_t)b * LSEQ + m16 * 16 + l15) * DDIM;
    const unsigned short* Trow[8];
    #pragma unroll
    for (int t = 0; t < 8; ++t)
        Trow[t] = Tb + ((size_t)b * LSEQ + nstrip * 128 + t * 16 + l15) * DDIM;

    v8f acc[8];
    #pragma unroll
    for (int t = 0; t < 8; ++t)
        acc[t] = (v8f){0.f, 0.f, 0.f, 0.f, 0.f, 0.f, 0.f, 0.f};

    for (int k0 = 0; k0 < DDIM; k0 += 32) {
        // ---- A tile: two b128 loads, already in WMMA register layout ----
        v16bf a;
        uint4* ap = reinterpret_cast<uint4*>(&a);
        ap[0] = *reinterpret_cast<const uint4*>(Srow + k0 + 8 * half);
        ap[1] = *reinterpret_cast<const uint4*>(Srow + k0 + 16 + 8 * half);
        // ---- 8 B tiles reuse the same A ----
        #pragma unroll
        for (int t = 0; t < 8; ++t) {
            v16bf bm;
            uint4* bp = reinterpret_cast<uint4*>(&bm);
            const unsigned short* p = Trow[t] + k0 + 8 * half;  // K = 8*half+0..7, 16+8*half+0..7
            bp[0] = *reinterpret_cast<const uint4*>(p);
            bp[1] = *reinterpret_cast<const uint4*>(p + 16);
            acc[t] = __builtin_amdgcn_wmma_f32_16x16x32_bf16(
                false, a, false, bm, (short)0, acc[t], false, false);
        }
    }

    // ---- fused distance epilogue ----
    float s2r[8];
    #pragma unroll
    for (int r = 0; r < 8; ++r)
        s2r[r] = s2[b * LSEQ + m16 * 16 + 8 * half + r];
    #pragma unroll
    for (int t = 0; t < 8; ++t) {
        const int n = nstrip * 128 + t * 16 + l15;
        const float t2v = t2[b * LSEQ + n];
        #pragma unroll
        for (int r = 0; r < 8; ++r) {
            const int l = m16 * 16 + 8 * half + r;
            const float sq = s2r[r] + t2v - 2.0f * acc[t][r];
            dist[((size_t)b * LSEQ + l) * LSEQ + n] = sqrtf(fmaxf(sq, 1e-12f));
        }
    }
}

// ---------------------------------------------------------------------------
// Stage 3: soft-DTW wavefront (gamma = 1). One block per batch, thread t owns
// DP column j = t+1. 3-buffer LDS rotation -> single barrier per diagonal.
// R boundary: R[0][0]=0, R[0][j>0]=R[i>0][0]=+inf. Softmin is inf-safe via
// min-shift (at least one predecessor of every live cell is finite).
// ---------------------------------------------------------------------------
__global__ __launch_bounds__(LSEQ) void softdtw_kernel(
    const float* __restrict__ dist, float* __restrict__ partial) {
    const int b = blockIdx.x;
    const int t = threadIdx.x;           // column j = t+1
    __shared__ float buf[3][LSEQ];
    const float INF = __builtin_inff();
    buf[0][t] = INF;
    buf[1][t] = INF;
    buf[2][t] = INF;
    float my_p1 = INF;                   // this thread's value from step k-1
    __syncthreads();

    int cur = 2, pm1 = 1, pm2 = 0;       // buffers: step k, k-1, k-2
    const float* db = dist + (size_t)b * LSEQ * LSEQ;

    for (int k = 2; k <= 2 * LSEQ; ++k) {
        const int i = k - (t + 1);       // DP row for this thread at diag k
        const bool active = (i >= 1) && (i <= LSEQ);
        if (active) {
            const float up   = my_p1;                                  // R[i-1][j]
            const float left = (t > 0) ? buf[pm1][t - 1] : INF;        // R[i][j-1]
            const float diag = (t > 0) ? buf[pm2][t - 1]
                                       : ((i == 1) ? 0.0f : INF);      // R[i-1][j-1]
            const float d = db[(size_t)(i - 1) * LSEQ + t];
            const float m = fminf(up, fminf(left, diag));              // finite
            const float s = __expf(m - up) + __expf(m - left) + __expf(m - diag);
            const float r_new = d + (m - __logf(s));                   // gamma=1
            buf[cur][t] = r_new;
            my_p1 = r_new;
        }
        __syncthreads();                 // writes@k visible before reads@k+1
        const int old = pm2; pm2 = pm1; pm1 = cur; cur = old;
    }
    if (t == LSEQ - 1) partial[b] = my_p1;   // R[LS][LT]
}

// ---------------------------------------------------------------------------
// Stage 4: mean over batches.
// ---------------------------------------------------------------------------
__global__ __launch_bounds__(BATCH) void mean_kernel(
    const float* __restrict__ partial, float* __restrict__ out) {
    __shared__ float sh[BATCH];
    const int t = threadIdx.x;
    sh[t] = partial[t];
    __syncthreads();
    if (t == 0) {
        float s = 0.0f;
        #pragma unroll
        for (int i = 0; i < BATCH; ++i) s += sh[i];
        out[0] = s / (float)BATCH;
    }
}

// ---------------------------------------------------------------------------
extern "C" void kernel_launch(void* const* d_in, const int* in_sizes, int n_in,
                              void* d_out, int out_size, void* d_ws, size_t ws_size,
                              hipStream_t stream) {
    const float* S = (const float*)d_in[0];   // (B, LS, D) f32
    const float* T = (const float*)d_in[1];   // (B, LT, D) f32

    // Workspace layout (16B-aligned blocks):
    //   dist    : B*L*L f32            (16 MB)
    //   Sb, Tb  : B*L*D bf16 each      (32 MB each)
    //   s2, t2  : B*L f32 each
    //   partial : B f32
    char* ws = (char*)d_ws;
    float*          dist    = (float*)ws;
    unsigned short* Sb      = (unsigned short*)(ws + (size_t)BATCH * LSEQ * LSEQ * 4);
    unsigned short* Tb      = Sb + (size_t)BATCH * LSEQ * DDIM;
    float*          s2      = (float*)(Tb + (size_t)BATCH * LSEQ * DDIM);
    float*          t2      = s2 + BATCH * LSEQ;
    float*          partial = t2 + BATCH * LSEQ;

    // Stage 1: 2*B*L rows, 8 rows (waves) per 256-thread block.
    convert_norms_kernel<<<dim3((2 * BATCH * LSEQ) / 8), 256, 0, stream>>>(
        S, T, Sb, Tb, s2, t2);

    // Stage 2: 32 wave-strips per batch, 8 waves per block -> (4, B).
    dist_wmma_kernel<<<dim3(4, BATCH), 256, 0, stream>>>(Sb, Tb, s2, t2, dist);

    // Stage 3: one workgroup per batch.
    softdtw_kernel<<<dim3(BATCH), LSEQ, 0, stream>>>(dist, partial);

    // Stage 4: scalar output.
    mean_kernel<<<dim3(1), BATCH, 0, stream>>>(partial, (float*)d_out);
}
